// NeuralODE_84198538871535
// MI455X (gfx1250) — matrix operand order repaired and checked by reading, not compiled, not run on hardware
//
#include <hip/hip_runtime.h>
#include <hip/hip_bf16.h>

typedef __attribute__((ext_vector_type(16))) _Float16 v16h;
typedef __attribute__((ext_vector_type(8)))  _Float16 v8h;
typedef __attribute__((ext_vector_type(8)))  float    v8f;

constexpr int B_ = 16384;
constexpr int D_ = 64;
constexpr int H_ = 256;
constexpr int O_ = 32;
constexpr int T_ = 50;
constexpr int NSUB = 8;

constexpr int WAVES = 4;                 // waves per block
constexpr int NT    = WAVES * 32;        // 128 threads
constexpr int MT    = 2;                 // 16-row M-tiles per wave
constexpr int ROWS_PER_WAVE  = 16 * MT;  // 32
constexpr int ROWS_PER_BLOCK = WAVES * ROWS_PER_WAVE;  // 128

union FragH { v16h v; v8h h[2]; };

// A-matrix 16x32 f16 fragment (ISA 7.12.2): lane l: m=l&15, hi=l>>4
//   h[0..7]  = A[m][kbase + 8*hi + j],  h[8..15] = A[m][kbase + 16 + 8*hi + j]
static __device__ __forceinline__ v16h load_a_frag(const _Float16* base, int stride,
                                                   int row, int kbase, int hi) {
  const _Float16* p = base + row * stride + kbase + 8 * hi;
  FragH f;
  f.h[0] = *(const v8h*)(p);
  f.h[1] = *(const v8h*)(p + 16);
  return f.v;
}

// B-matrix 32x16 f16 fragment: lane l: n=l&15, hi=l>>4
//   h[0..7] = B[kbase + 8*hi + j][n],  h[8..15] = B[kbase + 16 + 8*hi + j][n]
// Loaded from a transposed (NxK) LDS copy -> two 16-byte chunks per lane.
static __device__ __forceinline__ v16h load_b_frag(const _Float16* baseT, int stride,
                                                   int row, int kbase, int hi) {
  const _Float16* p = baseT + row * stride + kbase + 8 * hi;
  FragH f;
  f.h[0] = *(const v8h*)(p);
  f.h[1] = *(const v8h*)(p + 16);
  return f.v;
}

static __device__ __forceinline__ v8f wmma_f16(v16h a, v16h b, v8f c) {
  return __builtin_amdgcn_wmma_f32_16x16x32_f16(false, a, false, b, (short)0, c,
                                                false, false);
}

// tanh(x) = 1 - 2/(exp2(2x*log2e)+1); v_exp_f32 + v_rcp_f32, saturates correctly.
static __device__ __forceinline__ float fast_tanh(float x) {
  float e = __builtin_amdgcn_exp2f(x * 2.88539008177792681472f);
  return 1.0f - 2.0f * __builtin_amdgcn_rcpf(e + 1.0f);
}

__global__ __launch_bounds__(NT, 1)
void NeuralODE_kernel(const float* __restrict__ y0, const float* __restrict__ tarr,
                      const float* __restrict__ W1, const float* __restrict__ b1,
                      const float* __restrict__ W2, const float* __restrict__ b2,
                      const float* __restrict__ Wfc, const float* __restrict__ bfc,
                      float* __restrict__ out) {
  __shared__ __align__(16) _Float16 sW1T[H_ * D_];   // [h][d]  32 KB
  __shared__ __align__(16) _Float16 sW2T[D_ * H_];   // [d][h]  32 KB
  __shared__ __align__(16) _Float16 sWfcT[O_ * D_];  // [o][d]   4 KB
  __shared__ float sb1[H_];
  __shared__ float sb2[D_];
  __shared__ float sbfc[O_];
  // Per-wave staging: 32x64 f16 = 4 KB (hidden restage overlays first 2 KB).
  __shared__ __align__(16) _Float16 sStage[WAVES][ROWS_PER_WAVE * D_];

  const int tid  = threadIdx.x;
  const int wave = tid >> 5;
  const int lane = tid & 31;
  const int n    = lane & 15;   // tile column (also A-fragment row m)
  const int hi   = lane >> 4;   // lane-half select
  const int rowBase = blockIdx.x * ROWS_PER_BLOCK + wave * ROWS_PER_WAVE;
  _Float16* stg = &sStage[wave][0];

  // ---- stage weights into LDS (f16, transposed) ----
  for (int idx = tid; idx < D_ * H_; idx += NT) {
    int d = idx & (D_ - 1);
    int h = idx >> 6;
    sW1T[h * D_ + d] = (_Float16)W1[d * H_ + h];   // W1 is [D][H]
    sW2T[d * H_ + h] = (_Float16)W2[h * D_ + d];   // W2 is [H][D]
  }
  for (int idx = tid; idx < D_ * O_; idx += NT) {
    int o = idx & (O_ - 1);
    int d = idx >> 5;
    sWfcT[o * D_ + d] = (_Float16)Wfc[d * O_ + o]; // Wfc is [D][O]
  }
  for (int idx = tid; idx < H_; idx += NT) sb1[idx] = b1[idx];
  if (tid < D_) sb2[tid] = b2[tid];
  if (tid < O_) sbfc[tid] = bfc[tid];
  __syncthreads();

  // ---- state in C layout: y[mt][t][r] = Y[rowBase+16*mt + r+8*hi][16*t + n] ----
  v8f y[MT][4], k[MT][4], acc[MT][4];
#pragma unroll
  for (int mt = 0; mt < MT; ++mt)
#pragma unroll
    for (int t = 0; t < 4; ++t)
#pragma unroll
      for (int r = 0; r < 8; ++r)
        y[mt][t][r] = y0[(size_t)(rowBase + 16 * mt + r + 8 * hi) * D_ + 16 * t + n];

  // k <- f(y + c*k) = tanh((y+c*k)@W1 + b1) @ W2 + b2 ; fused hidden chunks of 32.
  auto feval = [&](float c) {
    // stage input as f16 [32][64] (fused y + c*k, no f32 temp)
#pragma unroll
    for (int mt = 0; mt < MT; ++mt)
#pragma unroll
      for (int t = 0; t < 4; ++t)
#pragma unroll
        for (int r = 0; r < 8; ++r)
          stg[(16 * mt + r + 8 * hi) * D_ + 16 * t + n] =
              (_Float16)(y[mt][t][r] + c * k[mt][t][r]);
    v16h a[MT][2];
#pragma unroll
    for (int mt = 0; mt < MT; ++mt) {
      a[mt][0] = load_a_frag(stg, D_, 16 * mt + n, 0, hi);
      a[mt][1] = load_a_frag(stg, D_, 16 * mt + n, 32, hi);
    }
    // init output accumulators (reuse k) with b2
#pragma unroll
    for (int mt = 0; mt < MT; ++mt)
#pragma unroll
      for (int nt = 0; nt < 4; ++nt) {
        float bv = sb2[16 * nt + n];
#pragma unroll
        for (int r = 0; r < 8; ++r) k[mt][nt][r] = bv;
      }
    for (int j = 0; j < 8; ++j) {            // hidden chunks of 32
      const int hb = 32 * j;
      v8f hc[MT][2];
#pragma unroll
      for (int tt = 0; tt < 2; ++tt) {       // weight frags shared across MT tiles
        float bv = sb1[hb + 16 * tt + n];
        v16h bw0 = load_b_frag(sW1T, D_, hb + 16 * tt + n, 0, hi);
        v16h bw1 = load_b_frag(sW1T, D_, hb + 16 * tt + n, 32, hi);
#pragma unroll
        for (int mt = 0; mt < MT; ++mt) {
#pragma unroll
          for (int r = 0; r < 8; ++r) hc[mt][tt][r] = bv;
          hc[mt][tt] = wmma_f16(a[mt][0], bw0, hc[mt][tt]);
          hc[mt][tt] = wmma_f16(a[mt][1], bw1, hc[mt][tt]);
        }
      }
      // tanh -> restage as f16 [32][32] (a[][] already resident in VGPRs)
#pragma unroll
      for (int mt = 0; mt < MT; ++mt)
#pragma unroll
        for (int tt = 0; tt < 2; ++tt)
#pragma unroll
          for (int r = 0; r < 8; ++r)
            stg[(16 * mt + r + 8 * hi) * 32 + 16 * tt + n] =
                (_Float16)fast_tanh(hc[mt][tt][r]);
      v16h ah[MT];
#pragma unroll
      for (int mt = 0; mt < MT; ++mt)
        ah[mt] = load_a_frag(stg, 32, 16 * mt + n, 0, hi);
#pragma unroll
      for (int nt = 0; nt < 4; ++nt) {       // W2 frags shared across MT tiles
        v16h bw = load_b_frag(sW2T, H_, 16 * nt + n, hb, hi);
#pragma unroll
        for (int mt = 0; mt < MT; ++mt)
          k[mt][nt] = wmma_f16(ah[mt], bw, k[mt][nt]);
      }
    }
  };

  // projection: out[tIdx] = y @ Wfc + bfc
  auto project = [&](int tIdx) {
#pragma unroll
    for (int mt = 0; mt < MT; ++mt)
#pragma unroll
      for (int t = 0; t < 4; ++t)
#pragma unroll
        for (int r = 0; r < 8; ++r)
          stg[(16 * mt + r + 8 * hi) * D_ + 16 * t + n] = (_Float16)y[mt][t][r];
    v16h ay[MT][2];
#pragma unroll
    for (int mt = 0; mt < MT; ++mt) {
      ay[mt][0] = load_a_frag(stg, D_, 16 * mt + n, 0, hi);
      ay[mt][1] = load_a_frag(stg, D_, 16 * mt + n, 32, hi);
    }
#pragma unroll
    for (int ot = 0; ot < 2; ++ot) {
      v16h bf0 = load_b_frag(sWfcT, D_, 16 * ot + n, 0, hi);
      v16h bf1 = load_b_frag(sWfcT, D_, 16 * ot + n, 32, hi);
      float bv = sbfc[16 * ot + n];
#pragma unroll
      for (int mt = 0; mt < MT; ++mt) {
        v8f p;
#pragma unroll
        for (int r = 0; r < 8; ++r) p[r] = bv;
        p = wmma_f16(ay[mt][0], bf0, p);
        p = wmma_f16(ay[mt][1], bf1, p);
#pragma unroll
        for (int r = 0; r < 8; ++r)
          out[((size_t)tIdx * B_ + rowBase + 16 * mt + r + 8 * hi) * O_ +
              16 * ot + n] = p[r];
      }
    }
  };

  project(0);

  for (int iv = 0; iv < T_ - 1; ++iv) {
    const float h  = (tarr[iv + 1] - tarr[iv]) * (1.0f / NSUB);
    const float hh = 0.5f * h;
    const float h6 = h * (1.0f / 6.0f);
    for (int sub = 0; sub < NSUB; ++sub) {
#pragma unroll
      for (int mt = 0; mt < MT; ++mt)
#pragma unroll
        for (int t = 0; t < 4; ++t) { v8f z{}; k[mt][t] = z; acc[mt][t] = z; }
#pragma unroll 1
      for (int s = 0; s < 4; ++s) {
        const float c = (s == 0) ? 0.0f : ((s == 3) ? h : hh);
        feval(c);
        const float w = (s == 1 || s == 2) ? 2.0f : 1.0f;
#pragma unroll
        for (int mt = 0; mt < MT; ++mt)
#pragma unroll
          for (int t = 0; t < 4; ++t) acc[mt][t] += w * k[mt][t];
      }
#pragma unroll
      for (int mt = 0; mt < MT; ++mt)
#pragma unroll
        for (int t = 0; t < 4; ++t) y[mt][t] += h6 * acc[mt][t];
    }
    project(iv + 1);
  }
}

extern "C" void kernel_launch(void* const* d_in, const int* in_sizes, int n_in,
                              void* d_out, int out_size, void* d_ws, size_t ws_size,
                              hipStream_t stream) {
  (void)in_sizes; (void)n_in; (void)d_ws; (void)ws_size; (void)out_size;
  const float* y0  = (const float*)d_in[0];
  const float* t   = (const float*)d_in[1];
  const float* W1  = (const float*)d_in[2];
  const float* b1  = (const float*)d_in[3];
  const float* W2  = (const float*)d_in[4];
  const float* b2  = (const float*)d_in[5];
  const float* Wfc = (const float*)d_in[6];
  const float* bfc = (const float*)d_in[7];
  float* out = (float*)d_out;

  dim3 grid(B_ / ROWS_PER_BLOCK);  // 128 blocks
  dim3 block(NT);                  // 128 threads = 4 waves
  NeuralODE_kernel<<<grid, block, 0, stream>>>(y0, t, W1, b1, W2, b2, Wfc, bfc, out);
}
